// MHAWithRecency_9474697855500
// MI455X (gfx1250) — compile-verified, matrix operand order checked by reading, not run on hardware
//
#include <hip/hip_runtime.h>
#include <hip/hip_bf16.h>
#include <stdint.h>

#define B_ 2
#define N_ 2048
#define D_ 1024
#define H_ 16
#define HD_ 64

typedef __attribute__((ext_vector_type(16))) __bf16 v16bf;
typedef __attribute__((ext_vector_type(8)))  float  v8f;
typedef __attribute__((ext_vector_type(8)))  unsigned short v8us;
typedef __attribute__((ext_vector_type(4)))  unsigned int v4u;
typedef __attribute__((ext_vector_type(8)))  int v8i;
typedef __attribute__((ext_vector_type(4)))  int v4i;

// round-to-nearest-even f32 -> bf16
__device__ __forceinline__ unsigned short f2bf(float x){
  unsigned u = __builtin_bit_cast(unsigned, x);
  u += 0x7FFFu + ((u >> 16) & 1u);
  return (unsigned short)(u >> 16);
}

__device__ __forceinline__ v8f wmma_bf16(v16bf a, v16bf b, v8f c){
  return __builtin_amdgcn_wmma_f32_16x16x32_bf16(false, a, false, b, (short)0, c, false, false);
}

// Load a 16x32 bf16 A/B fragment from a row-major [16 x ld] tile.
// A: lane = M row; B: lane = N col (tile stored [N,K] row-major).
// lanes 0-15: K 0-7 (v0-3), 16-23 (v4-7); lanes 16-31: K 8-15, 24-31.
__device__ __forceinline__ v16bf load_frag(const unsigned short* p0, int ld){
  int lane = threadIdx.x & 31;
  int r = lane & 15;
  int g = (lane >> 4) & 1;
  const unsigned short* p = p0 + (size_t)r * ld + g * 8;
  v8us lo = *(const v8us*)p;
  v8us hi = *(const v8us*)(p + 16);
  v16bf f;
#pragma unroll
  for (int i = 0; i < 8; ++i){
    f[i]     = __builtin_bit_cast(__bf16, (unsigned short)lo[i]);
    f[i + 8] = __builtin_bit_cast(__bf16, (unsigned short)hi[i]);
  }
  return f;
}

// Tensor Data Mover: DMA a [64 rows x 32 K] bf16 tile (rows stride = 1024 elems)
// from global into LDS (packed [64][32]). D# per CDNA5 ISA 8.3/8.4.
__device__ __forceinline__ void tdm_load_tile(unsigned short* lds_dst,
                                              const unsigned short* gsrc,
                                              unsigned rows_total){
  unsigned lds_addr = (unsigned)(uintptr_t)(void*)lds_dst;   // low 32 bits of generic = LDS offset
  unsigned long long ga = (unsigned long long)(uintptr_t)gsrc;
  v4u g0;
  g0[0] = 1u;                                                // count=1
  g0[1] = lds_addr;                                          // bits 63:32
  g0[2] = (unsigned)ga;                                      // global_addr[31:0]
  g0[3] = (unsigned)((ga >> 32) & 0x1FFFFFFu) | (2u << 30);  // global_addr[56:32] | type=2
  v8i g1;
  g1[0] = 0x00010000;                                   // workgroup_mask=0, data_size=1 (2B)
  g1[1] = (int)(((unsigned)D_ & 0xFFFFu) << 16);        // tensor_dim0 = 1024 (low16)
  g1[2] = (int)((rows_total & 0xFFFFu) << 16);          // dim0 hi16 (=0) | tensor_dim1 lo16
  g1[3] = (int)((rows_total >> 16) | (32u << 16));      // tensor_dim1 hi16 | tile_dim0 = 32
  g1[4] = 64;                                           // tile_dim1 = 64, tile_dim2 = 0
  g1[5] = D_;                                           // tensor_dim0_stride = 1024 (lo32)
  g1[6] = 0;                                            // stride0 hi16 | stride1 lo16
  g1[7] = 0;                                            // stride1 hi
  v4i z = {0, 0, 0, 0};
#if __has_include(<hip/amd_detail/amd_gfx1250_TDM.h>)
  v8i z8 = {0, 0, 0, 0, 0, 0, 0, 0};
  __builtin_amdgcn_tensor_load_to_lds(g0, g1, z, z, z8, 0);
#else
  __builtin_amdgcn_tensor_load_to_lds(g0, g1, z, z, 0);
#endif
}

__global__ void k_f2bf(const float* __restrict__ in, unsigned short* __restrict__ out, int n){
  int i = blockIdx.x * blockDim.x + threadIdx.x;
  if (i < n) out[i] = f2bf(in[i]);
}

// QKV projection: qkv[m,e] = sum_d X[m,d]*Wqkv[e,d] + bqkv[e]
// Block tile 64(M) x 64(N); B tile TDM-staged in LDS, double buffered.
// Scatter to Q[b,h,n,hd], K[b,h,n,hd], Vt[b,h,hd,n] (bf16).
__global__ __launch_bounds__(128) void k_qkv(const unsigned short* __restrict__ Xb,
    const unsigned short* __restrict__ Wb, const float* __restrict__ bias,
    unsigned short* __restrict__ Q, unsigned short* __restrict__ K,
    unsigned short* __restrict__ Vt){
  __shared__ unsigned short sB[2][64 * 32];
  const int w = threadIdx.x >> 5;
  const int lane = threadIdx.x & 31;
  const int r16 = lane & 15, g = (lane >> 4) & 1;
  const int mblk = blockIdx.x & 63;          // 4096/64
  const int nblk = blockIdx.x >> 6;          // 3072/64 = 48
  const int m0 = mblk * 64 + w * 16;
  const int n0 = nblk * 64;
  const bool leader = (__builtin_amdgcn_readfirstlane(threadIdx.x >> 5) == 0);
  v8f acc[4] = {};
  const unsigned short* Arow = Xb + (size_t)m0 * D_;
  const unsigned short* Bbase = Wb + (size_t)n0 * D_;
  int cur = 0;
  if (leader) tdm_load_tile(&sB[0][0], Bbase, 3 * D_);
  for (int k = 0; k < D_; k += 32){
    if (leader){
      if (k + 32 < D_){
        tdm_load_tile(&sB[cur ^ 1][0], Bbase + (k + 32), 3 * D_);
        __builtin_amdgcn_s_wait_tensorcnt(1);
      } else {
        __builtin_amdgcn_s_wait_tensorcnt(0);
      }
    }
    __syncthreads();                          // B tile ready
    v16bf a = load_frag(Arow + k, D_);
#pragma unroll
    for (int f = 0; f < 4; ++f){
      v16bf bb = load_frag(&sB[cur][f * 16 * 32], 32);
      acc[f] = wmma_bf16(a, bb, acc[f]);
    }
    __syncthreads();                          // done reading before overwrite
    cur ^= 1;
  }
#pragma unroll
  for (int f = 0; f < 4; ++f){
    int e = n0 + f * 16 + r16;
    float bv = bias[e];
#pragma unroll
    for (int r = 0; r < 8; ++r){
      int m = m0 + r + 8 * g;
      int b = m >> 11;
      int n = m & (N_ - 1);
      unsigned short hv = f2bf(acc[f][r] + bv);
      if (e < D_){
        int h = e >> 6, d = e & 63;
        Q[(((size_t)(b * H_ + h)) * N_ + n) * HD_ + d] = hv;
      } else if (e < 2 * D_){
        int e2 = e - D_; int h = e2 >> 6, d = e2 & 63;
        K[(((size_t)(b * H_ + h)) * N_ + n) * HD_ + d] = hv;
      } else {
        int e2 = e - 2 * D_; int h = e2 >> 6, d = e2 & 63;
        Vt[(((size_t)(b * H_ + h)) * HD_ + d) * N_ + n] = hv;
      }
    }
  }
}

// Flash attention with recency bias. 1 block = 64 queries of one (b,h).
__global__ __launch_bounds__(128) void k_attn(const unsigned short* __restrict__ Q,
    const unsigned short* __restrict__ K, const unsigned short* __restrict__ Vt,
    const int* __restrict__ tIdx, const unsigned char* __restrict__ mask,
    const float* __restrict__ alpha, unsigned short* __restrict__ ctx){
  __shared__ int sh_t[N_];
  __shared__ unsigned short sh_p[4][16 * 32];
  const int w = threadIdx.x >> 5, lane = threadIdx.x & 31;
  const int r16 = lane & 15, g = (lane >> 4) & 1;
  const int blk = blockIdx.x;
  const int qt = blk & 31;
  const int h  = (blk >> 5) & (H_ - 1);
  const int b  = blk >> 9;
  const int q0 = qt * 64 + w * 16;

  for (int i = threadIdx.x; i < N_; i += 128) sh_t[i] = tIdx[b * N_ + i];
  __syncthreads();

  const float al = alpha[h];
  const float scale = 0.125f;            // 1/sqrt(64)
  const unsigned short* Qb = Q + (((size_t)(b * H_ + h)) * N_ + q0) * HD_;
  const unsigned short* Kb = K + ((size_t)(b * H_ + h)) * N_ * HD_;
  const unsigned short* Vb = Vt + ((size_t)(b * H_ + h)) * HD_ * N_;

  v16bf qa0 = load_frag(Qb, HD_);
  v16bf qa1 = load_frag(Qb + 32, HD_);

  int tq[8];
#pragma unroll
  for (int r = 0; r < 8; ++r) tq[r] = sh_t[q0 + r + 8 * g];

  float mrun[8], lrun[8];
  v8f o[4] = {};
#pragma unroll
  for (int r = 0; r < 8; ++r){ mrun[r] = -3.0e38f; lrun[r] = 0.0f; }

  for (int j0 = 0; j0 < N_; j0 += 32){
    v8f s0 = {}, s1 = {};
    {
      v16bf kb = load_frag(Kb + (size_t)j0 * HD_, HD_);
      v16bf kc = load_frag(Kb + (size_t)j0 * HD_ + 32, HD_);
      s0 = wmma_bf16(qa0, kb, s0);
      s0 = wmma_bf16(qa1, kc, s0);
      kb = load_frag(Kb + (size_t)(j0 + 16) * HD_, HD_);
      kc = load_frag(Kb + (size_t)(j0 + 16) * HD_ + 32, HD_);
      s1 = wmma_bf16(qa0, kb, s1);
      s1 = wmma_bf16(qa1, kc, s1);
    }
    const int tk0 = sh_t[j0 + r16];
    const int tk1 = sh_t[j0 + 16 + r16];
#pragma unroll
    for (int r = 0; r < 8; ++r){
      int qabs = q0 + r + 8 * g;
      const unsigned char* mrow = mask + ((size_t)b * N_ + qabs) * N_ + j0 + r16;
      int d0 = tq[r] - tk0; if (d0 < 0) d0 = 0;
      int d1 = tq[r] - tk1; if (d1 < 0) d1 = 0;
      float s0r = s0[r] * scale - al * (float)d0;
      float s1r = s1[r] * scale - al * (float)d1;
      if (mrow[0])  s0r = -3.0e38f;
      if (mrow[16]) s1r = -3.0e38f;
      float bm = fmaxf(s0r, s1r);
#pragma unroll
      for (int mm = 1; mm < 16; mm <<= 1) bm = fmaxf(bm, __shfl_xor(bm, mm, 32));
      float mnew = fmaxf(mrun[r], bm);
      float corr = __expf(mrun[r] - mnew);
      float p0 = __expf(s0r - mnew);
      float p1 = __expf(s1r - mnew);
      float ps = p0 + p1;
#pragma unroll
      for (int mm = 1; mm < 16; mm <<= 1) ps += __shfl_xor(ps, mm, 32);
      lrun[r] = lrun[r] * corr + ps;
      mrun[r] = mnew;
#pragma unroll
      for (int f = 0; f < 4; ++f) o[f][r] *= corr;
      sh_p[w][(r + 8 * g) * 32 + r16]      = f2bf(p0);
      sh_p[w][(r + 8 * g) * 32 + 16 + r16] = f2bf(p1);
    }
    v16bf pa = load_frag(&sh_p[w][0], 32);
#pragma unroll
    for (int f = 0; f < 4; ++f){
      v16bf vb = load_frag(Vb + (size_t)(f * 16) * N_ + j0, N_);
      o[f] = wmma_bf16(pa, vb, o[f]);
    }
  }
#pragma unroll
  for (int r = 0; r < 8; ++r){
    float inv = 1.0f / lrun[r];
    int qabs = q0 + r + 8 * g;
#pragma unroll
    for (int f = 0; f < 4; ++f){
      ctx[((size_t)(b * N_ + qabs)) * D_ + h * HD_ + f * 16 + r16] = f2bf(o[f][r] * inv);
    }
  }
}

// Output projection: out[m,e] = sum_d ctx[m,d]*Wo[e,d] + bo[e]  (f32 out)
// Same TDM-staged structure as k_qkv.
__global__ __launch_bounds__(128) void k_out(const unsigned short* __restrict__ Cb,
    const unsigned short* __restrict__ Wb, const float* __restrict__ bias,
    float* __restrict__ out){
  __shared__ unsigned short sB[2][64 * 32];
  const int w = threadIdx.x >> 5, lane = threadIdx.x & 31;
  const int r16 = lane & 15, g = (lane >> 4) & 1;
  const int mblk = blockIdx.x & 63;          // 4096/64
  const int nblk = blockIdx.x >> 6;          // 1024/64 = 16
  const int m0 = mblk * 64 + w * 16;
  const int n0 = nblk * 64;
  const bool leader = (__builtin_amdgcn_readfirstlane(threadIdx.x >> 5) == 0);
  v8f acc[4] = {};
  const unsigned short* Arow = Cb + (size_t)m0 * D_;
  const unsigned short* Bbase = Wb + (size_t)n0 * D_;
  int cur = 0;
  if (leader) tdm_load_tile(&sB[0][0], Bbase, D_);
  for (int k = 0; k < D_; k += 32){
    if (leader){
      if (k + 32 < D_){
        tdm_load_tile(&sB[cur ^ 1][0], Bbase + (k + 32), D_);
        __builtin_amdgcn_s_wait_tensorcnt(1);
      } else {
        __builtin_amdgcn_s_wait_tensorcnt(0);
      }
    }
    __syncthreads();
    v16bf a = load_frag(Arow + k, D_);
#pragma unroll
    for (int f = 0; f < 4; ++f){
      v16bf bb = load_frag(&sB[cur][f * 16 * 32], 32);
      acc[f] = wmma_bf16(a, bb, acc[f]);
    }
    __syncthreads();
    cur ^= 1;
  }
#pragma unroll
  for (int f = 0; f < 4; ++f){
    int e = n0 + f * 16 + r16;
    float bv = bias[e];
#pragma unroll
    for (int r = 0; r < 8; ++r){
      int m = m0 + r + 8 * g;
      out[(size_t)m * D_ + e] = acc[f][r] + bv;
    }
  }
}

extern "C" void kernel_launch(void* const* d_in, const int* in_sizes, int n_in,
                              void* d_out, int out_size, void* d_ws, size_t ws_size,
                              hipStream_t stream){
  const float* X            = (const float*)d_in[0];
  const int* t_idx          = (const int*)d_in[1];
  const unsigned char* mask = (const unsigned char*)d_in[2];
  const float* Wqkv         = (const float*)d_in[3];
  const float* bqkv         = (const float*)d_in[4];
  const float* Wo           = (const float*)d_in[5];
  const float* bo           = (const float*)d_in[6];
  const float* alpha        = (const float*)d_in[7];

  char* ws = (char*)d_ws;
  size_t off = 0;
  unsigned short* Xb  = (unsigned short*)(ws + off); off += (size_t)B_ * N_ * D_ * 2;
  unsigned short* Wqb = (unsigned short*)(ws + off); off += (size_t)3 * D_ * D_ * 2;
  unsigned short* Wob = (unsigned short*)(ws + off); off += (size_t)D_ * D_ * 2;
  unsigned short* Qd  = (unsigned short*)(ws + off); off += (size_t)B_ * N_ * D_ * 2;
  unsigned short* Kd  = (unsigned short*)(ws + off); off += (size_t)B_ * N_ * D_ * 2;
  unsigned short* Vtd = (unsigned short*)(ws + off); off += (size_t)B_ * N_ * D_ * 2;
  unsigned short* Ctx = (unsigned short*)(ws + off);

  int n1 = B_ * N_ * D_;
  k_f2bf<<<(n1 + 255) / 256, 256, 0, stream>>>(X, Xb, n1);
  int n2 = 3 * D_ * D_;
  k_f2bf<<<(n2 + 255) / 256, 256, 0, stream>>>(Wqkv, Wqb, n2);
  int n3 = D_ * D_;
  k_f2bf<<<(n3 + 255) / 256, 256, 0, stream>>>(Wo, Wob, n3);

  k_qkv<<<64 * 48, 128, 0, stream>>>(Xb, Wqb, bqkv, Qd, Kd, Vtd);
  k_attn<<<B_ * H_ * (N_ / 64), 128, 0, stream>>>(Qd, Kd, Vtd, t_idx, mask, alpha, Ctx);
  k_out<<<64 * 16, 128, 0, stream>>>(Ctx, Wob, bo, (float*)d_out);
}
